// MutualNet_32581621907793
// MI455X (gfx1250) — matrix operand order repaired and checked by reading, not compiled, not run on hardware
//
#include <hip/hip_runtime.h>
#include <hip/hip_bf16.h>

// ---------------- constants ----------------
constexpr int B = 8, C = 512, K = 8, N = 4096;     // N = H*W
constexpr size_t NTOT = (size_t)B * C * N;          // elements of one [B,C,H,W]
constexpr int CC = C * C;

// ---------------- types ----------------
typedef __attribute__((ext_vector_type(16))) __bf16 v16bf;
typedef __attribute__((ext_vector_type(8)))  __bf16 v8bf;
typedef __attribute__((ext_vector_type(8)))  float  v8f;
typedef __attribute__((ext_vector_type(4)))  unsigned u32x4;
typedef __attribute__((ext_vector_type(8)))  unsigned u32x8;

// ---------------- helpers ----------------
__device__ __forceinline__ unsigned f2bf(float f) {
    unsigned u = __float_as_uint(f);
    return (u + 0x7FFFu + ((u >> 16) & 1u)) >> 16;   // RNE to bf16
}

__device__ __forceinline__ float block_reduce(float v, float* red, int tid, int nthr) {
    red[tid] = v; __syncthreads();
    for (int s = nthr >> 1; s > 0; s >>= 1) {
        if (tid < s) red[tid] += red[tid + s];
        __syncthreads();
    }
    float r = red[0]; __syncthreads();
    return r;
}

// ============================================================
// GraphNet stage
// ============================================================
__global__ __launch_bounds__(C) void k_prep(const float* __restrict__ anchor,
                                            const float* __restrict__ sigma_p,
                                            float* __restrict__ inv2, float* __restrict__ winv,
                                            float* __restrict__ sig, float* __restrict__ ck) {
    __shared__ float red[C];
    int k = blockIdx.x, c = threadIdx.x;
    float sp = sigma_p[k * C + c];
    float s  = 1.f / (1.f + __expf(-sp));
    float iv = 1.f / (s * s);
    float a  = anchor[k * C + c];
    sig[k * C + c]  = s;
    inv2[k * C + c] = iv;
    winv[k * C + c] = a * iv;
    float t = block_reduce(a * a * iv, red, c, C);
    if (c == 0) ck[k] = t;
}

__global__ __launch_bounds__(256) void k_assign(const float* __restrict__ x,
                                                const float* __restrict__ inv2,
                                                const float* __restrict__ winv,
                                                const float* __restrict__ ck,
                                                float* __restrict__ sa) {
    __shared__ float sInv[K * C];
    __shared__ float sW[K * C];
    for (int t = threadIdx.x; t < K * C; t += 256) { sInv[t] = inv2[t]; sW[t] = winv[t]; }
    __syncthreads();
    int gid = blockIdx.x * 256 + threadIdx.x;     // over B*N
    int b = gid >> 12, n = gid & (N - 1);
    const float* xb = x + (size_t)b * C * N + n;
    float qa[K], qb[K];
#pragma unroll
    for (int k = 0; k < K; ++k) { qa[k] = 0.f; qb[k] = 0.f; }
    for (int c = 0; c < C; ++c) {
        float xv = xb[(size_t)c * N];
        float x2 = xv * xv;
#pragma unroll
        for (int k = 0; k < K; ++k) {
            qa[k] = fmaf(x2, sInv[k * C + c], qa[k]);
            qb[k] = fmaf(xv, sW[k * C + c], qb[k]);
        }
    }
    float lg[K], mx = -1e30f;
#pragma unroll
    for (int k = 0; k < K; ++k) {
        lg[k] = -0.5f * (qa[k] - 2.f * qb[k] + ck[k]);
        mx = fmaxf(mx, lg[k]);
    }
    float den = 0.f;
#pragma unroll
    for (int k = 0; k < K; ++k) { lg[k] = __expf(lg[k] - mx); den += lg[k]; }
    float inv = 1.f / den;
#pragma unroll
    for (int k = 0; k < K; ++k)
        sa[((size_t)b * K + k) * N + n] = lg[k] * inv;
}

__global__ __launch_bounds__(256) void k_colsum(const float* __restrict__ sa, float* __restrict__ s) {
    __shared__ float red[256];
    int bk = blockIdx.x;
    const float* p = sa + (size_t)bk * N;
    float acc = 0.f;
    for (int n = threadIdx.x; n < N; n += 256) acc += p[n];
    float t = block_reduce(acc, red, threadIdx.x, 256);
    if (threadIdx.x == 0) s[bk] = t;
}

__global__ __launch_bounds__(256) void k_num(const float* __restrict__ x,
                                             const float* __restrict__ sa,
                                             float* __restrict__ num) {
    int gtid = blockIdx.x * 256 + threadIdx.x;
    int wv = gtid >> 5, lane = gtid & 31;
    int b  = wv >> 7;             // C/4 = 128 waves per batch
    int c0 = (wv & 127) * 4;
    const float* xb  = x  + (size_t)b * C * N;
    const float* sab = sa + (size_t)b * K * N;
    float acc[4][K];
#pragma unroll
    for (int j = 0; j < 4; ++j)
#pragma unroll
        for (int k = 0; k < K; ++k) acc[j][k] = 0.f;
    for (int n = lane; n < N; n += 32) {
        float sv[K];
#pragma unroll
        for (int k = 0; k < K; ++k) sv[k] = sab[(size_t)k * N + n];
#pragma unroll
        for (int j = 0; j < 4; ++j) {
            float xv = xb[(size_t)(c0 + j) * N + n];
#pragma unroll
            for (int k = 0; k < K; ++k) acc[j][k] = fmaf(sv[k], xv, acc[j][k]);
        }
    }
#pragma unroll
    for (int j = 0; j < 4; ++j)
#pragma unroll
        for (int k = 0; k < K; ++k) {
            float v = acc[j][k];
            for (int off = 16; off >= 1; off >>= 1) v += __shfl_xor(v, off, 32);
            acc[j][k] = v;
        }
    if (lane == 0)
#pragma unroll
        for (int j = 0; j < 4; ++j)
#pragma unroll
            for (int k = 0; k < K; ++k)
                num[((size_t)b * K + k) * C + c0 + j] = acc[j][k];
}

__global__ __launch_bounds__(C) void k_nodes(const float* __restrict__ num,
                                             const float* __restrict__ s,
                                             const float* __restrict__ anchor,
                                             const float* __restrict__ sig,
                                             float* __restrict__ g) {
    __shared__ float red[C];
    __shared__ float knorm[K];
    int b = blockIdx.x, c = threadIdx.x;
    float v[K];
#pragma unroll
    for (int k = 0; k < K; ++k) {
        float sk = s[b * K + k];
        float nv = num[((size_t)b * K + k) * C + c] - sk * anchor[k * C + c];
        v[k] = nv / sig[k * C + c] / (sk + 1e-9f);
    }
    for (int k = 0; k < K; ++k) {
        float t = block_reduce(v[k] * v[k], red, c, C);
        if (c == 0) knorm[k] = sqrtf(t);
        __syncthreads();
    }
    float tot = 0.f;
#pragma unroll
    for (int k = 0; k < K; ++k) {
        v[k] = v[k] / fmaxf(knorm[k], 1e-12f);
        tot += v[k] * v[k];
    }
    float gn = sqrtf(block_reduce(tot, red, c, C));
    float ign = 1.f / fmaxf(gn, 1e-12f);
#pragma unroll
    for (int k = 0; k < K; ++k)
        g[(size_t)b * (K * C) + k * C + c] = v[k] * ign;
}

// ============================================================
// BasicConv on [B,C,8]
// ============================================================
__global__ __launch_bounds__(256) void k_bcmm(const float* __restrict__ g,
                                              const float* __restrict__ w,
                                              float* __restrict__ y) {
    __shared__ float sg[K * C];
    int b = blockIdx.y;
    for (int i = threadIdx.x; i < K * C; i += 256) sg[i] = g[(size_t)b * K * C + i];
    __syncthreads();
    int t = blockIdx.x * 256 + threadIdx.x;
    int o = t >> 3, l = t & 7;
    float acc = 0.f;
    for (int c = 0; c < C; ++c) acc = fmaf(w[(size_t)o * C + c], sg[c * K + l], acc);
    y[(size_t)b * K * C + t] = acc;
}

__global__ __launch_bounds__(C) void k_bn_small(const float* __restrict__ y,
                                                const float* __restrict__ gamma,
                                                const float* __restrict__ beta,
                                                float* __restrict__ out) {
    int o = threadIdx.x;
    float sum = 0.f, ssq = 0.f;
    for (int b = 0; b < B; ++b)
#pragma unroll
        for (int l = 0; l < K; ++l) {
            float v = y[(size_t)b * K * C + o * K + l];
            sum += v; ssq += v * v;
        }
    float mu  = sum * (1.f / (B * K));
    float var = ssq * (1.f / (B * K)) - mu * mu;
    float sc  = gamma[o] * rsqrtf(var + 1e-5f);
    float sh  = beta[o] - mu * sc;
    for (int b = 0; b < B; ++b)
#pragma unroll
        for (int l = 0; l < K; ++l) {
            size_t idx = (size_t)b * K * C + o * K + l;
            out[idx] = fmaxf(y[idx] * sc + sh, 0.f);
        }
}

// ============================================================
// Gram + softmax
// ============================================================
__global__ __launch_bounds__(64) void k_gram(const float* __restrict__ X,
                                             const float* __restrict__ Y,
                                             float* __restrict__ A) {
    __shared__ float sx[K * C];
    __shared__ float sy[K * C];
    __shared__ float G[64];
    int b = blockIdx.x, t = threadIdx.x;
    for (int i = t; i < K * C; i += 64) { sx[i] = X[(size_t)b * K * C + i]; sy[i] = Y[(size_t)b * K * C + i]; }
    __syncthreads();
    int k = t >> 3, l = t & 7;
    float acc = 0.f;
    for (int c = 0; c < C; ++c) acc = fmaf(sx[c * K + k], sy[c * K + l], acc);
    G[t] = acc; __syncthreads();
    if (l == 0) {
        float mx = -1e30f;
#pragma unroll
        for (int j = 0; j < K; ++j) mx = fmaxf(mx, G[k * 8 + j]);
        float den = 0.f; float e[K];
#pragma unroll
        for (int j = 0; j < K; ++j) { e[j] = __expf(G[k * 8 + j] - mx); den += e[j]; }
        float inv = 1.f / den;
#pragma unroll
        for (int j = 0; j < K; ++j) A[(size_t)b * 64 + k * 8 + j] = e[j] * inv;
    }
}

// mode: 0 plain, 1 relu, 2 relu then add addbuf
__global__ __launch_bounds__(128) void k_cstep(const float* __restrict__ x,
                                               const float* __restrict__ w,
                                               const float* __restrict__ adj,
                                               const float* __restrict__ addbuf,
                                               float* __restrict__ out, int mode) {
    __shared__ float sx[K * C];
    __shared__ float sadj[64];
    int b = blockIdx.y;
    for (int i = threadIdx.x; i < K * C; i += 128) sx[i] = x[(size_t)b * K * C + i];
    if (threadIdx.x < 64) sadj[threadIdx.x] = adj[(size_t)b * 64 + threadIdx.x];
    __syncthreads();
    int d = blockIdx.x * 128 + threadIdx.x;
    float sup[K];
#pragma unroll
    for (int l = 0; l < K; ++l) sup[l] = 0.f;
    for (int c = 0; c < C; ++c) {
        float wv = w[(size_t)c * C + d];
#pragma unroll
        for (int l = 0; l < K; ++l) sup[l] = fmaf(sx[c * K + l], wv, sup[l]);
    }
    float o[K];
#pragma unroll
    for (int k = 0; k < K; ++k) {
        float a = 0.f;
#pragma unroll
        for (int l = 0; l < K; ++l) a = fmaf(sadj[k * 8 + l], sup[l], a);
        o[k] = a;
    }
    if (mode >= 1)
#pragma unroll
        for (int k = 0; k < K; ++k) o[k] = fmaxf(o[k], 0.f);
    if (mode == 2)
#pragma unroll
        for (int k = 0; k < K; ++k) o[k] += addbuf[(size_t)b * K * C + d * K + k];
#pragma unroll
    for (int k = 0; k < K; ++k) out[(size_t)b * K * C + d * K + k] = o[k];
}

__global__ __launch_bounds__(256) void k_mix(const float* __restrict__ A,
                                             const float* __restrict__ g2b,
                                             float* __restrict__ m) {
    __shared__ float sA[64];
    int b = blockIdx.y;
    if (threadIdx.x < 64) sA[threadIdx.x] = A[(size_t)b * 64 + threadIdx.x];
    __syncthreads();
    int t = blockIdx.x * 256 + threadIdx.x;
    int c = t >> 3, k = t & 7;
    float acc = 0.f;
#pragma unroll
    for (int l = 0; l < K; ++l) acc = fmaf(sA[k * 8 + l], g2b[(size_t)b * K * C + c * K + l], acc);
    m[(size_t)b * K * C + t] = acc;
}

// ============================================================
// Back-projection to pixels in bf16
// ============================================================
__global__ __launch_bounds__(256) void k_backproj(const float* __restrict__ X,
                                                  const float* __restrict__ sa,
                                                  unsigned short* __restrict__ Pbf) {
    int bc = blockIdx.x;
    int b = bc >> 9, c = bc & (C - 1);
    float xv[K];
#pragma unroll
    for (int k = 0; k < K; ++k) xv[k] = X[(size_t)b * K * C + c * K + k];
    const float* sab = sa + (size_t)b * K * N;
    unsigned short* dst = Pbf + ((size_t)b * C + c) * N;
    for (int it = 0; it < 8; ++it) {
        int n = (it * 256 + threadIdx.x) * 2;
        float a0 = 0.f, a1 = 0.f;
#pragma unroll
        for (int k = 0; k < K; ++k) {
            a0 = fmaf(xv[k], sab[(size_t)k * N + n], a0);
            a1 = fmaf(xv[k], sab[(size_t)k * N + n + 1], a1);
        }
        unsigned pkd = f2bf(a0) | (f2bf(a1) << 16);
        *(unsigned*)(dst + n) = pkd;
    }
}

__global__ __launch_bounds__(256) void k_w2bf(const float* __restrict__ w,
                                              unsigned short* __restrict__ wbf) {
    int i = (blockIdx.x * 256 + threadIdx.x) * 2;
    unsigned pkd = f2bf(w[i]) | (f2bf(w[i + 1]) << 16);
    *(unsigned*)(wbf + i) = pkd;
}

// ============================================================
// Big GEMM:  Y[b,o,n] = sum_c W[o,c] * P[b,c,n]
//  - A panel (64 x 512 bf16, contiguous rows) DMA'd once per block by the
//    Tensor Data Mover (tensor_load_to_lds, TENSORcnt-tracked)
//  - B tiles double-buffered in LDS; one barrier per K-step
//  - 8 waves: 4(M) x 2(N); wave tile 16x64; v_wmma_f32_16x16x32_bf16
// ============================================================
#define GTM 64
#define GTN 128
#define GTK 32

__device__ __forceinline__ void stage_b(const unsigned short* __restrict__ Pb,
                                        int k0, int n0, int tid,
                                        unsigned short* __restrict__ dst) {
    int c = tid >> 3, np = (tid & 7) * 16;
    const unsigned short* src = Pb + (size_t)(k0 + c) * N + n0 + np;
    uint4 q0 = *(const uint4*)(src);
    uint4 q1 = *(const uint4*)(src + 8);
    unsigned vv[8] = {q0.x, q0.y, q0.z, q0.w, q1.x, q1.y, q1.z, q1.w};
#pragma unroll
    for (int j = 0; j < 8; ++j) {
        dst[(np + 2 * j) * GTK + c]     = (unsigned short)(vv[j] & 0xFFFFu);
        dst[(np + 2 * j + 1) * GTK + c] = (unsigned short)(vv[j] >> 16);
    }
}

__global__ __launch_bounds__(256) void k_gemm(const unsigned short* __restrict__ Wbf,
                                              const unsigned short* __restrict__ Pbf,
                                              float* __restrict__ Y) {
    __shared__ __align__(16) unsigned short sA[GTM * C];          // 64 KB: [64 rows][512 k]
    __shared__ __align__(16) unsigned short sB[2][GTN * GTK];     // 2 x 8 KB: [128 n][32 k]
    int b  = blockIdx.z;
    int m0 = blockIdx.y * GTM;
    int n0 = blockIdx.x * GTN;
    int tid = threadIdx.x, wave = tid >> 5, lane = tid & 31;
    int wm0 = (wave >> 1) * 16;
    int wn0 = (wave & 1) * 64;
    int r  = lane & 15;
    int hf = lane >> 4;

    // ---- TDM: DMA the whole A panel (rows m0..m0+63, all 512 k) into LDS ----
    if (wave == 0) {
        unsigned long long ga = (unsigned long long)Wbf + (unsigned long long)m0 * (C * 2);
        unsigned lds0 = (unsigned)(unsigned long long)(void*)&sA[0];
        u32x4 g0;
        g0[0] = 1u;                                              // count=1, no gather
        g0[1] = lds0;                                            // lds_addr
        g0[2] = (unsigned)ga;                                    // global_addr[31:0]
        g0[3] = ((unsigned)(ga >> 32) & 0x01FFFFFFu) | 0x80000000u; // addr[56:32] | type=2
        u32x8 g1;
        g1[0] = 0x00010000u;                                     // data_size=2B, mask=0
        g1[1] = (unsigned)C << 16;                               // tensor_dim0 = 512
        g1[2] = (unsigned)C << 16;                               // tensor_dim1 = 512
        g1[3] = (unsigned)C << 16;                               // tile_dim0 = 512
        g1[4] = (unsigned)GTM;                                   // tile_dim1 = 64
        g1[5] = (unsigned)C;                                     // tensor_dim0_stride = 512
        g1[6] = 0u;
        g1[7] = 0u;
        asm volatile("tensor_load_to_lds %0, %1" : : "s"(g0), "s"(g1) : "memory");
        __builtin_amdgcn_s_wait_tensorcnt(0);
    }

    const unsigned short* Pb = Pbf + (size_t)b * C * N;
    stage_b(Pb, 0, n0, tid, sB[0]);                              // prologue B tile
    __syncthreads();

    v8f acc[4];
#pragma unroll
    for (int s = 0; s < 4; ++s)
#pragma unroll
        for (int j = 0; j < 8; ++j) acc[s][j] = 0.f;

    for (int i = 0; i < C / GTK; ++i) {
        int k0 = i * GTK;
        const unsigned short* curB = sB[i & 1];

        // ---- A fragment (ISA 16-bit 16x32 lane layout), panel row stride = 512 ----
        v16bf afrag;
        {
            const v8bf* p = (const v8bf*)(sA + (wm0 + r) * C + k0);
            v8bf lo = p[hf];       // lanes<16: K0..7  ; lanes>=16: K8..15
            v8bf hi = p[hf + 2];   // lanes<16: K16..23; lanes>=16: K24..31
#pragma unroll
            for (int j = 0; j < 8; ++j) { afrag[j] = lo[j]; afrag[8 + j] = hi[j]; }
        }
        // ---- preload all 4 B fragments, then back-to-back WMMAs ----
        v16bf bfrag[4];
#pragma unroll
        for (int s = 0; s < 4; ++s) {
            int ncol = wn0 + s * 16 + r;
            const v8bf* p = (const v8bf*)(curB + ncol * GTK);
            v8bf blo = p[hf];
            v8bf bhi = p[hf + 2];
#pragma unroll
            for (int j = 0; j < 8; ++j) { bfrag[s][j] = blo[j]; bfrag[s][8 + j] = bhi[j]; }
        }
#pragma unroll
        for (int s = 0; s < 4; ++s)
            acc[s] = __builtin_amdgcn_wmma_f32_16x16x32_bf16(
                false, afrag, false, bfrag[s], (short)0, acc[s], false, false);

        // ---- stage next B tile into the other buffer ----
        if (i + 1 < C / GTK) {
            stage_b(Pb, k0 + GTK, n0, tid, sB[(i + 1) & 1]);
            if (i + 2 < C / GTK)
                __builtin_prefetch((const void*)(Pb + (size_t)(k0 + 2 * GTK + (tid >> 3)) * N
                                                 + n0 + (tid & 7) * 16), 0, 1);
        }
        __syncthreads();
    }

    // ---- epilogue: lane r = column, VGPR v -> row v + 8*hf ----
    float* Yb = Y + (size_t)b * C * N;
#pragma unroll
    for (int s = 0; s < 4; ++s) {
        int n = n0 + wn0 + s * 16 + r;
#pragma unroll
        for (int v = 0; v < 8; ++v) {
            int m = m0 + wm0 + v + 8 * hf;
            Yb[(size_t)m * N + n] = acc[s][v];
        }
    }
}

// ============================================================
// BatchNorm over (b,n) + ReLU + residual add (in place on Y)
// ============================================================
__global__ __launch_bounds__(256) void k_bnstats(const float* __restrict__ Y,
                                                 const float* __restrict__ gamma,
                                                 const float* __restrict__ beta,
                                                 float* __restrict__ sc, float* __restrict__ sh) {
    __shared__ float red[256];
    int o = blockIdx.x;
    float sum = 0.f, ssq = 0.f;
    for (int i = threadIdx.x; i < B * N; i += 256) {
        int b = i >> 12, n = i & (N - 1);
        float v = Y[((size_t)b * C + o) * N + n];
        sum += v; ssq += v * v;
    }
    float ts = block_reduce(sum, red, threadIdx.x, 256);
    float tq = block_reduce(ssq, red, threadIdx.x, 256);
    if (threadIdx.x == 0) {
        float mu  = ts * (1.f / (B * N));
        float var = tq * (1.f / (B * N)) - mu * mu;
        float s   = gamma[o] * rsqrtf(var + 1e-5f);
        sc[o] = s;
        sh[o] = beta[o] - mu * s;
    }
}

__global__ __launch_bounds__(256) void k_apply(const float* __restrict__ xorig,
                                               const float* __restrict__ sc,
                                               const float* __restrict__ sh,
                                               float* __restrict__ Y) {
    size_t i = (size_t)blockIdx.x * 256 + threadIdx.x;
    int o = (int)((i >> 12) & (C - 1));
    float v = Y[i] * sc[o] + sh[o];
    Y[i] = xorig[i] + fmaxf(v, 0.f);
}

// ============================================================
// host orchestration
// ============================================================
extern "C" void kernel_launch(void* const* d_in, const int* in_sizes, int n_in,
                              void* d_out, int out_size, void* d_ws, size_t ws_size,
                              hipStream_t stream) {
    (void)in_sizes; (void)n_in; (void)out_size; (void)ws_size;
    const float* x1      = (const float*)d_in[0];
    const float* x2      = (const float*)d_in[1];
    const float* anchors = (const float*)d_in[2];
    const float* sigmas  = (const float*)d_in[3];
    const float* conv_w  = (const float*)d_in[4];
    const float* conv_g  = (const float*)d_in[5];
    const float* conv_b  = (const float*)d_in[6];
    const float* gcn_w   = (const float*)d_in[7];
    const float* mconv_w = (const float*)d_in[8];
    const float* mconv_g = (const float*)d_in[9];
    const float* mconv_b = (const float*)d_in[10];
    float* out = (float*)d_out;

    char* wsb = (char*)d_ws; size_t off = 0;
    auto alloc = [&](size_t bytes) -> void* {
        void* p = wsb + off;
        off += (bytes + 255) & ~(size_t)255;
        return p;
    };

    float* saA  = (float*)alloc((size_t)B * K * N * 4);
    float* saB  = (float*)alloc((size_t)B * K * N * 4);
    float* inv2b = (float*)alloc(K * C * 4);
    float* winvb = (float*)alloc(K * C * 4);
    float* sigb  = (float*)alloc(K * C * 4);
    float* ckb   = (float*)alloc(K * 4);
    float* sb    = (float*)alloc(B * K * 4);
    float* numb  = (float*)alloc((size_t)B * K * C * 4);
    float* g1    = (float*)alloc((size_t)B * K * C * 4);
    float* g2    = (float*)alloc((size_t)B * K * C * 4);
    float* bco[6];
    for (int j = 0; j < 6; ++j) bco[j] = (float*)alloc((size_t)B * K * C * 4);
    float* ypre  = (float*)alloc((size_t)B * K * C * 4);
    float* Amat  = (float*)alloc((size_t)B * 64 * 4);
    float* mbuf  = (float*)alloc((size_t)B * K * C * 4);
    float* casct = (float*)alloc((size_t)B * K * C * 4);
    float* xab   = (float*)alloc((size_t)B * K * C * 4);
    float* xbb   = (float*)alloc((size_t)B * K * C * 4);
    float* scb   = (float*)alloc(C * 4);
    float* shb   = (float*)alloc(C * 4);
    unsigned short* Wbf = (unsigned short*)alloc((size_t)CC * 2);
    unsigned short* Pbf = (unsigned short*)alloc(NTOT * 2);

    // -------- GraphNet for x1 and x2 --------
    for (int i = 0; i < 2; ++i) {
        const float* xin = i ? x2 : x1;
        float* saP = i ? saB : saA;
        float* g   = i ? g2 : g1;
        k_prep  <<<dim3(K),           dim3(C),   0, stream>>>(anchors + i * K * C, sigmas + i * K * C,
                                                              inv2b, winvb, sigb, ckb);
        k_assign<<<dim3(B * N / 256), dim3(256), 0, stream>>>(xin, inv2b, winvb, ckb, saP);
        k_colsum<<<dim3(B * K),       dim3(256), 0, stream>>>(saP, sb);
        k_num   <<<dim3(B * C / 32),  dim3(256), 0, stream>>>(xin, saP, numb);
        k_nodes <<<dim3(B),           dim3(C),   0, stream>>>(numb, sb, anchors + i * K * C, sigb, g);
    }

    // -------- six BasicConvs --------
    for (int j = 0; j < 6; ++j) {
        const float* gin = (j == 0 || j >= 4) ? g1 : g2;
        k_bcmm    <<<dim3(16, B), dim3(256), 0, stream>>>(gin, conv_w + (size_t)j * CC, ypre);
        k_bn_small<<<dim3(1),     dim3(C),   0, stream>>>(ypre, conv_g + j * C, conv_b + j * C, bco[j]);
    }

    auto cascade = [&](const float* xin, const float* wA, const float* wB,
                       const float* addb, float* xout) {
        k_gram <<<dim3(B), dim3(64), 0, stream>>>(xin, xin, Amat);
        k_cstep<<<dim3(4, B), dim3(128), 0, stream>>>(xin, wA, Amat, (const float*)nullptr, casct, 0);
        k_gram <<<dim3(B), dim3(64), 0, stream>>>(casct, casct, Amat);
        k_cstep<<<dim3(4, B), dim3(128), 0, stream>>>(casct, wB, Amat, addb, xout, addb ? 2 : 1);
    };

    auto mutual = [&](const float* g1a, const float* g2a, const float* g2b,
                      const float* assign, const float* gwbase,
                      const float* cw, const float* cg, const float* cb,
                      const float* xorig, float* yout) {
        k_gram<<<dim3(B), dim3(64), 0, stream>>>(g1a, g2a, Amat);
        k_mix <<<dim3(16, B), dim3(256), 0, stream>>>(Amat, g2b, mbuf);
        cascade(g2b, gwbase + 0 * (size_t)CC, gwbase + 1 * (size_t)CC, mbuf, xab);
        cascade(xab, gwbase + 2 * (size_t)CC, gwbase + 3 * (size_t)CC, (const float*)nullptr, xbb);
        cascade(xbb, gwbase + 4 * (size_t)CC, gwbase + 5 * (size_t)CC, (const float*)nullptr, xab);
        k_backproj<<<dim3(B * C), dim3(256), 0, stream>>>(xab, assign, Pbf);
        k_w2bf    <<<dim3(CC / 512), dim3(256), 0, stream>>>(cw, Wbf);
        k_gemm    <<<dim3(N / GTN, C / GTM, B), dim3(256), 0, stream>>>(Wbf, Pbf, yout);
        k_bnstats <<<dim3(C), dim3(256), 0, stream>>>(yout, cg, cb, scb, shb);
        k_apply   <<<dim3((unsigned)(NTOT / 256)), dim3(256), 0, stream>>>(xorig, scb, shb, yout);
    };

    // mutual 0 -> x2_out (second half of d_out)
    mutual(bco[0], bco[1], bco[2], saB, gcn_w,
           mconv_w, mconv_g, mconv_b, x2, out + NTOT);
    // mutual 1 -> x1_out (first half of d_out)
    mutual(bco[3], bco[4], bco[5], saA, gcn_w + 6 * (size_t)CC,
           mconv_w + (size_t)CC, mconv_g + C, mconv_b + C, x1, out);
}
